// AttentionPropagationLayer_83021717831844
// MI455X (gfx1250) — matrix operand order repaired and checked by reading, not compiled, not run on hardware
//
#include <hip/hip_runtime.h>
#include <hip/hip_bf16.h>

// ---------------------------------------------------------------------------
// CDNA5 / gfx1250 implementation: bf16 WMMA (16x16x32, f32 acc) for both MLPs.
// wave32; 8 waves per block; each wave owns a 16-row stripe of a 128-row tile.
// Scatter-add uses native fire-and-forget global_atomic_add_f32 @ device scope.
// ---------------------------------------------------------------------------

typedef __attribute__((ext_vector_type(16))) __bf16          v16bf;
typedef __attribute__((ext_vector_type(8)))  float           v8f;
typedef __attribute__((ext_vector_type(16))) unsigned short  v16u16;
typedef __attribute__((ext_vector_type(8)))  unsigned short  v8u16;

// float -> bf16 round-to-nearest-even
__device__ __forceinline__ unsigned short f2bf(float f) {
  unsigned int u = __float_as_uint(f);
  u = (u + 0x7FFFu + ((u >> 16) & 1u)) >> 16;
  return (unsigned short)u;
}

__device__ __forceinline__ v8f wmma_bf16(v16bf a, v16bf b, v8f c) {
  // D = A(16x32 bf16) * B(32x16 bf16) + C(16x16 f32)
  return __builtin_amdgcn_wmma_f32_16x16x32_bf16(
      /*neg_a=*/false, a, /*neg_b=*/false, b,
      /*c_mod=*/(short)0, c, /*reuse_a=*/false, /*reuse_b=*/false);
}

// A-fragment from LDS (row-major bf16, stride sx ushorts).
// ISA 16-bit A 16x32 layout: lane l -> row (l&15), K-half kb=(l>>4)*8;
// u[0..7]=K k0+kb+0..7 ; u[8..15]=K k0+kb+16..23  -> two 16B ds loads.
__device__ __forceinline__ v16bf load_a(const unsigned short* X, int sx,
                                        int rowBase, int k0, int lane) {
  int m  = lane & 15;
  int kb = (lane >> 4) << 3;
  const unsigned short* p = X + (rowBase + m) * sx + k0 + kb;
  v8u16 lo = *(const v8u16*)p;
  v8u16 hi = *(const v8u16*)(p + 16);
  v16u16 u = __builtin_shufflevector(lo, hi, 0,1,2,3,4,5,6,7,8,9,10,11,12,13,14,15);
  return __builtin_bit_cast(v16bf, u);
}

// B-fragment: weights are pre-packed so each lane's 16 values are contiguous.
__device__ __forceinline__ v16bf load_b(const unsigned short* p) {
  v16u16 u = *(const v16u16*)p;
  return __builtin_bit_cast(v16bf, u);
}

// Native non-returning FP32 add atomic at device scope (L2-side RMW).
// Avoids any CAS-loop expansion; tracked with STOREcnt (fire-and-forget).
__device__ __forceinline__ void atomic_add_f32(float* p, float v) {
  asm volatile("global_atomic_add_f32 %0, %1, off scope:SCOPE_DEV"
               :
               : "v"(p), "v"(v)
               : "memory");
}

// ---------------------------------------------------------------------------
// Pack f32 weight [K][N] into bf16 WMMA-B fragment order:
// out[((ct*(K/32)+ks)*32 + lane)*16 + t] = W[ks*32 + (lane>>4)*8 + (t&7) + (t>>3)*16][ct*16 + (lane&15)]
// ---------------------------------------------------------------------------
__global__ void pack_w_kernel(const float* __restrict__ W,
                              unsigned short* __restrict__ out, int K, int N) {
  int gid = blockIdx.x * 256 + threadIdx.x;
  if (gid >= K * N) return;
  int t    = gid & 15;
  int lane = (gid >> 4) & 31;
  int tk   = gid >> 9;
  int KS   = K >> 5;
  int ks   = tk % KS;
  int ct   = tk / KS;
  int k    = ks * 32 + ((lane >> 4) << 3) + (t & 7) + ((t >> 3) << 4);
  int n    = ct * 16 + (lane & 15);
  out[gid] = f2bf(W[(size_t)k * N + n]);
}

// ---------------------------------------------------------------------------
// Hidden GEMM layer: LDS X (128 x KDIM bf16) @ packed W (KDIM x NOUT) + bias,
// optional relu, result to LDS (bf16).  Wave `wave` owns rows [16w, 16w+16).
// Two output tiles in flight per K-loop for WMMA back-to-back issue.
// ---------------------------------------------------------------------------
template <int KDIM, int NOUT, bool RELU>
__device__ __forceinline__ void gemm_hidden(const unsigned short* Xl, int sx,
                                            const unsigned short* __restrict__ Wpk,
                                            const float* __restrict__ bias,
                                            unsigned short* Hl, int sh,
                                            int wave, int lane) {
  const int rowBase = wave << 4;
  constexpr int KS = KDIM / 32;
  for (int ct = 0; ct < NOUT / 16; ct += 2) {
    v8f c0 = {}, c1 = {};
    for (int ks = 0; ks < KS; ++ks) {
      v16bf a  = load_a(Xl, sx, rowBase, ks * 32, lane);
      v16bf b0 = load_b(Wpk + ((size_t)((ct * KS + ks) * 32 + lane) << 4));
      v16bf b1 = load_b(Wpk + ((size_t)(((ct + 1) * KS + ks) * 32 + lane) << 4));
      c0 = wmma_bf16(a, b0, c0);
      c1 = wmma_bf16(a, b1, c1);
    }
    int n  = lane & 15;
    int mo = (lane >> 4) << 3;   // C layout: VGPR j -> M = j + mo
    float bi0 = bias[ct * 16 + n];
    float bi1 = bias[ct * 16 + 16 + n];
#pragma unroll
    for (int j = 0; j < 8; ++j) {
      int m = rowBase + j + mo;
      float v0 = c0[j] + bi0;
      float v1 = c1[j] + bi1;
      if (RELU) { v0 = fmaxf(v0, 0.0f); v1 = fmaxf(v1, 0.0f); }
      Hl[m * sh + ct * 16 + n]      = f2bf(v0);
      Hl[m * sh + ct * 16 + 16 + n] = f2bf(v1);
    }
  }
}

// ---------------------------------------------------------------------------
// Kernel 1: message MLP (gather -> 320->256->256->128 -> atomic scatter-add)
// Block = 128 edges.  LDS: X(128x328) | H1(128x264) | verts(128x2); H2 aliases X.
// ---------------------------------------------------------------------------
__global__ void __launch_bounds__(256) msg_kernel(
    const float* __restrict__ ns, const float* __restrict__ eg,
    const int* __restrict__ vt,
    const unsigned short* __restrict__ W1, const float* __restrict__ b1,
    const unsigned short* __restrict__ W2, const float* __restrict__ b2,
    const unsigned short* __restrict__ W3, const float* __restrict__ b3,
    float* __restrict__ summed) {
  extern __shared__ __align__(16) unsigned short smem[];
  const int SX = 328;   // 320 + pad (164 dwords; conflict-free b128 A loads)
  const int SH = 264;   // 256 + pad (132 dwords)
  unsigned short* Xl = smem;                     // also reused as H2
  unsigned short* H1 = smem + 128 * SX;
  int* vl = (int*)(smem + 128 * SX + 128 * SH);  // 128 x 2 vertex ids

  const int eBase = blockIdx.x * 128;
  const int tid = threadIdx.x;

  vl[tid] = vt[eBase * 2 + tid];                 // 256 ints, 256 threads
  __syncthreads();

  // Build X = [ns[v0] | ns[v1] | edge] as bf16
  for (int idx = tid; idx < 128 * 256; idx += 256) {
    int r = idx >> 8, c = idx & 255;
    int v = vl[(r << 1) + (c >> 7)];
    Xl[r * SX + c] = f2bf(ns[((size_t)v << 7) + (c & 127)]);
  }
  for (int idx = tid; idx < 128 * 64; idx += 256) {
    int r = idx >> 6, c = idx & 63;
    Xl[r * SX + 256 + c] = f2bf(eg[(((size_t)(eBase + r)) << 6) + c]);
  }
  __syncthreads();

  const int wave = tid >> 5, lane = tid & 31;
  gemm_hidden<320, 256, true>(Xl, SX, W1, b1, H1, SH, wave, lane);
  __syncthreads();
  gemm_hidden<256, 256, true>(H1, SH, W2, b2, Xl, SH, wave, lane);  // H2 -> X region
  __syncthreads();

  // Layer 3 (256 -> 128) fused with dual-endpoint atomic scatter-add.
  const int rowBase = wave << 4;
  for (int ct = 0; ct < 8; ct += 2) {
    v8f c0 = {}, c1 = {};
    for (int ks = 0; ks < 8; ++ks) {
      v16bf a  = load_a(Xl, SH, rowBase, ks * 32, lane);
      v16bf t0 = load_b(W3 + ((size_t)((ct * 8 + ks) * 32 + lane) << 4));
      v16bf t1 = load_b(W3 + ((size_t)(((ct + 1) * 8 + ks) * 32 + lane) << 4));
      c0 = wmma_bf16(a, t0, c0);
      c1 = wmma_bf16(a, t1, c1);
    }
    int n  = lane & 15;
    int mo = (lane >> 4) << 3;
    float bi0 = b3[ct * 16 + n];
    float bi1 = b3[ct * 16 + 16 + n];
#pragma unroll
    for (int j = 0; j < 8; ++j) {
      int m = rowBase + j + mo;
      float v0 = c0[j] + bi0;
      float v1 = c1[j] + bi1;
      int d0 = vl[2 * m], d1 = vl[2 * m + 1];
      atomic_add_f32(&summed[((size_t)d0 << 7) + ct * 16 + n], v0);
      atomic_add_f32(&summed[((size_t)d1 << 7) + ct * 16 + n], v0);
      atomic_add_f32(&summed[((size_t)d0 << 7) + ct * 16 + 16 + n], v1);
      atomic_add_f32(&summed[((size_t)d1 << 7) + ct * 16 + 16 + n], v1);
    }
  }
}

// ---------------------------------------------------------------------------
// Kernel 2: update MLP.  X = [ns[n] | summed[n] | ns[n]-ns[n^2048]] (384)
// -> 256 -> 256 -> 128 -> d_out (f32).  Block = 128 nodes.
// ---------------------------------------------------------------------------
__global__ void __launch_bounds__(256) upd_kernel(
    const float* __restrict__ ns, const float* __restrict__ summed,
    const unsigned short* __restrict__ W1, const float* __restrict__ b1,
    const unsigned short* __restrict__ W2, const float* __restrict__ b2,
    const unsigned short* __restrict__ W3, const float* __restrict__ b3,
    float* __restrict__ out) {
  extern __shared__ __align__(16) unsigned short smem2[];
  const int SX = 392;   // 384 + pad (196 dwords; conflict-free)
  const int SH = 264;
  unsigned short* Xl = smem2;                    // also reused as H2
  unsigned short* H1 = smem2 + 128 * SX;

  const int nBase = blockIdx.x * 128;
  const int tid = threadIdx.x;

  for (int idx = tid; idx < 128 * 128; idx += 256) {
    int r = idx >> 7, c = idx & 127;
    int gn = nBase + r;
    float s  = ns[((size_t)gn << 7) + c];
    float p  = ns[((size_t)(gn ^ 2048) << 7) + c];   // attention partner
    float sm = summed[((size_t)gn << 7) + c];
    Xl[r * SX + c]       = f2bf(s);
    Xl[r * SX + 128 + c] = f2bf(sm);
    Xl[r * SX + 256 + c] = f2bf(s - p);
  }
  __syncthreads();

  const int wave = tid >> 5, lane = tid & 31;
  gemm_hidden<384, 256, true>(Xl, SX, W1, b1, H1, SH, wave, lane);
  __syncthreads();
  gemm_hidden<256, 256, true>(H1, SH, W2, b2, Xl, SH, wave, lane);  // H2 -> X region
  __syncthreads();

  const int rowBase = wave << 4;
  for (int ct = 0; ct < 8; ct += 2) {
    v8f c0 = {}, c1 = {};
    for (int ks = 0; ks < 8; ++ks) {
      v16bf a  = load_a(Xl, SH, rowBase, ks * 32, lane);
      v16bf t0 = load_b(W3 + ((size_t)((ct * 8 + ks) * 32 + lane) << 4));
      v16bf t1 = load_b(W3 + ((size_t)(((ct + 1) * 8 + ks) * 32 + lane) << 4));
      c0 = wmma_bf16(a, t0, c0);
      c1 = wmma_bf16(a, t1, c1);
    }
    int n  = lane & 15;
    int mo = (lane >> 4) << 3;
    float bi0 = b3[ct * 16 + n];
    float bi1 = b3[ct * 16 + 16 + n];
#pragma unroll
    for (int j = 0; j < 8; ++j) {
      int m = rowBase + j + mo;
      out[((size_t)(nBase + m) << 7) + ct * 16 + n]      = c0[j] + bi0;
      out[((size_t)(nBase + m) << 7) + ct * 16 + 16 + n] = c1[j] + bi1;
    }
  }
}

// ---------------------------------------------------------------------------
// Launch
// ---------------------------------------------------------------------------
extern "C" void kernel_launch(void* const* d_in, const int* in_sizes, int n_in,
                              void* d_out, int out_size, void* d_ws, size_t ws_size,
                              hipStream_t stream) {
  const float* ns  = (const float*)d_in[0];
  const float* eg  = (const float*)d_in[1];
  const int*   vt  = (const int*)d_in[2];
  const float* mW1 = (const float*)d_in[3];  const float* mb1 = (const float*)d_in[4];
  const float* mW2 = (const float*)d_in[5];  const float* mb2 = (const float*)d_in[6];
  const float* mW3 = (const float*)d_in[7];  const float* mb3 = (const float*)d_in[8];
  const float* uW1 = (const float*)d_in[9];  const float* ub1 = (const float*)d_in[10];
  const float* uW2 = (const float*)d_in[11]; const float* ub2 = (const float*)d_in[12];
  const float* uW3 = (const float*)d_in[13]; const float* ub3 = (const float*)d_in[14];
  float* out = (float*)d_out;

  const int N = 65536, E = 524288;

  // Workspace: summed (N x 128 f32) + bf16 packed weights.
  char* ws = (char*)d_ws;
  float* summed = (float*)ws;
  size_t off = (size_t)N * 128 * 4;
  unsigned short* w1m = (unsigned short*)(ws + off); off += (size_t)320 * 256 * 2;
  unsigned short* w2m = (unsigned short*)(ws + off); off += (size_t)256 * 256 * 2;
  unsigned short* w3m = (unsigned short*)(ws + off); off += (size_t)256 * 128 * 2;
  unsigned short* w1u = (unsigned short*)(ws + off); off += (size_t)384 * 256 * 2;
  unsigned short* w2u = (unsigned short*)(ws + off); off += (size_t)256 * 256 * 2;
  unsigned short* w3u = (unsigned short*)(ws + off); off += (size_t)256 * 128 * 2;

  hipMemsetAsync(summed, 0, (size_t)N * 128 * 4, stream);

  pack_w_kernel<<<(320 * 256) / 256, 256, 0, stream>>>(mW1, w1m, 320, 256);
  pack_w_kernel<<<(256 * 256) / 256, 256, 0, stream>>>(mW2, w2m, 256, 256);
  pack_w_kernel<<<(256 * 128) / 256, 256, 0, stream>>>(mW3, w3m, 256, 128);
  pack_w_kernel<<<(384 * 256) / 256, 256, 0, stream>>>(uW1, w1u, 384, 256);
  pack_w_kernel<<<(256 * 256) / 256, 256, 0, stream>>>(uW2, w2u, 256, 256);
  pack_w_kernel<<<(256 * 128) / 256, 256, 0, stream>>>(uW3, w3u, 256, 128);

  const size_t shm_msg = (size_t)(128 * 328 + 128 * 264) * 2 + 128 * 2 * 4; // 152576 B
  const size_t shm_upd = (size_t)(128 * 392 + 128 * 264) * 2;               // 167936 B

  msg_kernel<<<E / 128, 256, shm_msg, stream>>>(ns, eg, vt, w1m, mb1, w2m, mb2,
                                                w3m, mb3, summed);
  upd_kernel<<<N / 128, 256, shm_upd, stream>>>(ns, summed, w1u, ub1, w2u, ub2,
                                                w3u, ub3, out);
}
